// SelfAttentionWithClustering_6055903888020
// MI455X (gfx1250) — matrix operand order repaired and checked by reading, not compiled, not run on hardware
//
#include <hip/hip_runtime.h>
#include <hip/hip_bf16.h>
#include <stdint.h>

// ---------------- problem constants ----------------
#define BB 8
#define SS 1024
#define EE 1024
#define HH 16
#define HD 64          // head dim
#define E3 (3*EE)

typedef unsigned short u16;
typedef __attribute__((ext_vector_type(16))) __bf16 v16bf;
typedef __attribute__((ext_vector_type(8)))  float  v8f;

union FragBF {
    v16bf v;
    u16   u[16];
    uint4 q[2];
};

__device__ __forceinline__ u16 f2bf(float f) {
    unsigned u = __float_as_uint(f);
    u += 0x7fffu + ((u >> 16) & 1u);      // round-to-nearest-even
    return (u16)(u >> 16);
}

__device__ __forceinline__ v8f wmma_bf16(const v16bf& a, const v16bf& b, const v8f& c) {
    return __builtin_amdgcn_wmma_f32_16x16x32_bf16(false, a, false, b, (short)0, c, false, false);
}

// ---------------- CDNA5 async global->LDS copy (16B) ----------------
#if defined(__has_builtin)
#  if __has_builtin(__builtin_amdgcn_global_load_async_to_lds_b128)
#    define HAVE_ASYNC_LDS 1
#  endif
#endif
#ifndef HAVE_ASYNC_LDS
#  define HAVE_ASYNC_LDS 0
#endif

#if HAVE_ASYNC_LDS
#  if defined(__has_builtin) && __has_builtin(__builtin_amdgcn_s_wait_asynccnt)
#    define WAIT_ASYNC() __builtin_amdgcn_s_wait_asynccnt(0)
#  else
#    define WAIT_ASYNC() asm volatile("s_wait_asynccnt 0x0" ::: "memory")
#  endif
#else
#  define WAIT_ASYNC() ((void)0)
#endif

// exact parameter element type per hipcc diagnostic:
// 'int __attribute__((vector_size(4*sizeof(int)))) __device__ *'
typedef int vsi4 __attribute__((vector_size(16)));
typedef __attribute__((address_space(1))) vsi4* as1_v4i_p;
typedef __attribute__((address_space(3))) vsi4* as3_v4i_p;

__device__ __forceinline__ void async_cp16(const u16* g, u16* l) {
#if HAVE_ASYNC_LDS
    // generic global pointer bits == AS1 pointer bits; generic LDS pointer low
    // 32 bits == AS3 offset (aperture lives in bits 63:32), so integer casts
    // are exact and avoid addrspacecast portability issues.
    __builtin_amdgcn_global_load_async_to_lds_b128(
        (as1_v4i_p)(unsigned long long)(uintptr_t)g,
        (as3_v4i_p)(unsigned)(uintptr_t)l,
        0, 0);
#else
    *(uint4*)l = *(const uint4*)g;
#endif
}

// ---------------- elementwise helpers ----------------
__global__ void f32_to_bf16_kernel(const float* __restrict__ in, u16* __restrict__ out, int n) {
    int i = blockIdx.x * 256 + threadIdx.x;
    if (i < n) out[i] = f2bf(in[i]);
}

// v part of qkv [B,S,3E] (bf16) -> vT [B,H,HD,S] (bf16), key index contiguous
__global__ void transpose_v_kernel(const u16* __restrict__ qkv, u16* __restrict__ vT) {
    int idx = blockIdx.x * 256 + threadIdx.x;          // B*H*HD*S = 8,388,608
    if (idx >= BB * HH * HD * SS) return;
    int s  = idx & (SS - 1);
    int dd = (idx >> 10) & (HD - 1);
    int bh = idx >> 16;
    int b = bh >> 4, h = bh & 15;
    vT[idx] = qkv[(size_t)(b * SS + s) * E3 + 2 * EE + h * HD + dd];
}

// ---------------- WMMA GEMM:  out[M,N] = A[M,K] @ Bm[N,K]^T + bias ----------------
// Block: 256 threads (8 waves), block tile 128x128, wave tile 32x64, K-step 32.
// A/B tiles double-buffered in LDS via async global->LDS copies.
#define LSTRIDE 48   // LDS row stride in halves (96B, 16B aligned, conflict-reducing)

__global__ void wmma_gemm_bf16_kernel(const u16* __restrict__ A, const u16* __restrict__ Bm,
                                      const float* __restrict__ bias, void* __restrict__ out,
                                      int M, int N, int K, int store_bf16) {
    __shared__ __align__(16) u16 Abuf[2][128 * LSTRIDE];
    __shared__ __align__(16) u16 Bbuf[2][128 * LSTRIDE];

    int tid  = threadIdx.x;
    int lane = tid & 31;
    int wave = tid >> 5;
    int hs   = (lane >> 4) & 1;
    int mt   = (wave & 3) * 32;          // wave's M offset in block tile
    int nh   = (wave >> 2) * 64;         // wave's N offset in block tile
    int m0 = blockIdx.x * 128;
    int n0 = blockIdx.y * 128;

    // staging: 512 16B-chunks each for A and B; each thread copies 2 of each.
    int c0 = tid * 2;
    auto stage = [&](int buf, int k0) {
#pragma unroll
        for (int i = 0; i < 2; ++i) {
            int c = c0 + i;
            int row = c >> 2;
            int off = (c & 3) * 8;       // halves
            async_cp16(A  + (size_t)(m0 + row) * K + k0 + off, &Abuf[buf][row * LSTRIDE + off]);
            async_cp16(Bm + (size_t)(n0 + row) * K + k0 + off, &Bbuf[buf][row * LSTRIDE + off]);
        }
    };

    v8f acc[2][4] = {};
    stage(0, 0);
    for (int k0 = 0; k0 < K; k0 += 32) {
        int cur = (k0 >> 5) & 1;
        WAIT_ASYNC();                    // own stage of `cur` complete
        __syncthreads();                 // all waves staged `cur`; all done reading `cur^1`
        if (k0 + 32 < K) stage(cur ^ 1, k0 + 32);

        FragBF fa[2], fb;
#pragma unroll
        for (int t = 0; t < 2; ++t) {
            const u16* ar = &Abuf[cur][(mt + t * 16 + (lane & 15)) * LSTRIDE + (hs ? 8 : 0)];
            fa[t].q[0] = *(const uint4*)(ar);
            fa[t].q[1] = *(const uint4*)(ar + 16);
        }
#pragma unroll
        for (int nt = 0; nt < 4; ++nt) {
            const u16* br = &Bbuf[cur][(nh + nt * 16 + (lane & 15)) * LSTRIDE + (hs ? 8 : 0)];
            fb.q[0] = *(const uint4*)(br);
            fb.q[1] = *(const uint4*)(br + 16);
            acc[0][nt] = wmma_bf16(fa[0].v, fb.v, acc[0][nt]);
            acc[1][nt] = wmma_bf16(fa[1].v, fb.v, acc[1][nt]);
        }
    }

#pragma unroll
    for (int t = 0; t < 2; ++t) {
#pragma unroll
        for (int r = 0; r < 8; ++r) {
            int row = m0 + mt + t * 16 + hs * 8 + r;
#pragma unroll
            for (int nt = 0; nt < 4; ++nt) {
                int col = n0 + nh + nt * 16 + (lane & 15);
                float val = acc[t][nt][r] + bias[col];
                size_t o = (size_t)row * N + col;
                if (store_bf16) ((u16*)out)[o] = f2bf(val);
                else            ((float*)out)[o] = val;
            }
        }
    }
}

// ---------------- fused attention ----------------
// grid.x = B*(S/16): one block per (batch, 16-row q tile). 256 threads = 8 waves.
// Wave w processes heads w and w+8. Head-average accumulated in LDS (ds_add_f32),
// then stored once (no global atomics).
__global__ void attention_kernel(const u16* __restrict__ qkv, const u16* __restrict__ vT,
                                 float* __restrict__ attnw, u16* __restrict__ obuf) {
    __shared__ float wsum[16 * SS];                    // 64KB head-sum of probs
    __shared__ __align__(16) u16 pbuf[8][16 * 32];     // per-wave p staging
    int lane = threadIdx.x & 31;
    int wave = threadIdx.x >> 5;
    int hs   = (lane >> 4) & 1;
    int b  = blockIdx.x >> 6;
    int qt = blockIdx.x & 63;
    int qrow0 = qt * 16;
    const float scale = 0.125f;                        // 1/sqrt(64)

    for (int i = threadIdx.x; i < 16 * SS; i += 256) wsum[i] = 0.0f;
    __syncthreads();

    u16* myp = pbuf[wave];

    for (int hh = 0; hh < 2; ++hh) {
        int h  = wave + hh * 8;
        int bh = b * HH + h;

        // q A-fragments (K=64 -> two k-steps of 32)
        FragBF qf0, qf1;
        const u16* qrow = qkv + (size_t)(b * SS + qrow0 + (lane & 15)) * E3 + h * HD + (hs ? 8 : 0);
        qf0.q[0] = *(const uint4*)(qrow);      qf0.q[1] = *(const uint4*)(qrow + 16);
        qf1.q[0] = *(const uint4*)(qrow + 32); qf1.q[1] = *(const uint4*)(qrow + 48);

        // ---- pass 1: online row max / sum ----
        float mrun[8], lrun[8];
#pragma unroll
        for (int r = 0; r < 8; ++r) { mrun[r] = -1e30f; lrun[r] = 0.0f; }

        for (int n0 = 0; n0 < SS; n0 += 16) {
            FragBF kf0, kf1;
            const u16* krow = qkv + (size_t)(b * SS + n0 + (lane & 15)) * E3 + EE + h * HD + (hs ? 8 : 0);
            kf0.q[0] = *(const uint4*)(krow);      kf0.q[1] = *(const uint4*)(krow + 16);
            kf1.q[0] = *(const uint4*)(krow + 32); kf1.q[1] = *(const uint4*)(krow + 48);
            v8f c = {};
            c = wmma_bf16(qf0.v, kf0.v, c);
            c = wmma_bf16(qf1.v, kf1.v, c);
#pragma unroll
            for (int r = 0; r < 8; ++r) {
                float s = c[r] * scale;
                float t = s;
                for (int m = 1; m < 16; m <<= 1) t = fmaxf(t, __shfl_xor(t, m, 32));
                float mnew = fmaxf(mrun[r], t);
                float p = __expf(s - mnew);
                float ps = p;
                for (int m = 1; m < 16; m <<= 1) ps += __shfl_xor(ps, m, 32);
                lrun[r] = lrun[r] * __expf(mrun[r] - mnew) + ps;
                mrun[r] = mnew;
            }
        }
        float invl[8];
#pragma unroll
        for (int r = 0; r < 8; ++r) invl[r] = 1.0f / lrun[r];

        // ---- pass 2: normalize, accumulate o = p @ v, head-sum into LDS ----
        v8f oacc[4] = {};
        for (int kt = 0; kt < SS; kt += 32) {
#pragma unroll
            for (int sub = 0; sub < 2; ++sub) {
                int n0 = kt + sub * 16;
                FragBF kf0, kf1;
                const u16* krow = qkv + (size_t)(b * SS + n0 + (lane & 15)) * E3 + EE + h * HD + (hs ? 8 : 0);
                kf0.q[0] = *(const uint4*)(krow);      kf0.q[1] = *(const uint4*)(krow + 16);
                kf1.q[0] = *(const uint4*)(krow + 32); kf1.q[1] = *(const uint4*)(krow + 48);
                v8f c = {};
                c = wmma_bf16(qf0.v, kf0.v, c);
                c = wmma_bf16(qf1.v, kf1.v, c);
#pragma unroll
                for (int r = 0; r < 8; ++r) {
                    float p = __expf(c[r] * scale - mrun[r]) * invl[r];
                    atomicAdd(&wsum[(hs * 8 + r) * SS + n0 + (lane & 15)], p);   // ds_add_f32
                    myp[(hs * 8 + r) * 32 + sub * 16 + (lane & 15)] = f2bf(p);
                }
            }
            asm volatile("s_wait_dscnt 0x0" ::: "memory");   // wave-private LDS staging
            FragBF pf;
            const u16* prow = myp + (lane & 15) * 32 + (hs ? 8 : 0);
            pf.q[0] = *(const uint4*)(prow);
            pf.q[1] = *(const uint4*)(prow + 16);
#pragma unroll
            for (int nt = 0; nt < 4; ++nt) {
                FragBF vf;
                const u16* vrow = vT + (size_t)(bh * HD + nt * 16 + (lane & 15)) * SS + kt + (hs ? 8 : 0);
                vf.q[0] = *(const uint4*)(vrow);
                vf.q[1] = *(const uint4*)(vrow + 16);
                oacc[nt] = wmma_bf16(pf.v, vf.v, oacc[nt]);
            }
        }

        // o epilogue (bf16, [B,S,E] layout for out-projection GEMM)
#pragma unroll
        for (int r = 0; r < 8; ++r) {
            int srow = qrow0 + hs * 8 + r;
#pragma unroll
            for (int nt = 0; nt < 4; ++nt) {
                int col = h * HD + nt * 16 + (lane & 15);
                obuf[(size_t)(b * SS + srow) * EE + col] = f2bf(oacc[nt][r]);
            }
        }
    }

    __syncthreads();
    // single coalesced store of head average (replaces 134M global atomics)
    const float inv = 1.0f / HH;
    float* dst = attnw + (size_t)b * SS * SS + (size_t)qrow0 * SS;
    for (int i = threadIdx.x; i < 16 * SS; i += 256) dst[i] = wsum[i] * inv;
}

// ---------------- sequential greedy clustering (one block per batch) ----------------
__global__ void cluster_kernel(const float* __restrict__ attnw,
                               float* __restrict__ cidOut, int* __restrict__ covered) {
    __shared__ int visited[SS];
    int b = blockIdx.x;
    int j = threadIdx.x;
    visited[j] = 0;
    int cid = -1;
    __syncthreads();
    const float* aw = attnw + (size_t)b * SS * SS;
    for (int i = 0; i < SS; ++i) {
        int process = (visited[i] == 0);
        float a = aw[(size_t)i * SS + j];
        int claimed = process & (a > 0.5f);
        int newly = claimed & (visited[j] == 0);
        if (newly) cid = i;
        __syncthreads();
        if (claimed) visited[j] = 1;
        __syncthreads();
    }
    cidOut[b * SS + j] = (float)cid;
    covered[b * SS + j] = visited[j];
}

__global__ void clustered_out_kernel(const float* __restrict__ x, const int* __restrict__ covered,
                                     float* __restrict__ out) {
    int idx = blockIdx.x * 256 + threadIdx.x;
    if (idx >= BB * SS * EE) return;
    int tok = idx >> 10;
    out[idx] = covered[tok] ? x[idx] : -1e9f;
}

// ---------------- launcher ----------------
extern "C" void kernel_launch(void* const* d_in, const int* in_sizes, int n_in,
                              void* d_out, int out_size, void* d_ws, size_t ws_size,
                              hipStream_t stream) {
    (void)in_sizes; (void)n_in; (void)out_size; (void)ws_size;
    const float* x     = (const float*)d_in[0];   // [B,S,E]
    const float* w_in  = (const float*)d_in[1];   // [3E,E]
    const float* b_in  = (const float*)d_in[2];   // [3E]
    const float* w_out = (const float*)d_in[3];   // [E,E]
    const float* b_out = (const float*)d_in[4];   // [E]

    float* out          = (float*)d_out;
    float* attn_output  = out;                                      // B*S*E
    float* attn_weights = out + (size_t)BB * SS * EE;               // B*S*S
    float* clustered    = attn_weights + (size_t)BB * SS * SS;      // B*S*E
    float* cids         = clustered + (size_t)BB * SS * EE;         // B*S

    size_t off = 0;
    auto wsAlloc = [&](size_t bytes) -> char* {
        char* p = (char*)d_ws + off;
        off += (bytes + 255) & ~(size_t)255;
        return p;
    };
    u16* x_bf     = (u16*)wsAlloc((size_t)BB * SS * EE * 2);
    u16* w_in_bf  = (u16*)wsAlloc((size_t)E3 * EE * 2);
    u16* w_out_bf = (u16*)wsAlloc((size_t)EE * EE * 2);
    u16* qkv_bf   = (u16*)wsAlloc((size_t)BB * SS * E3 * 2);
    u16* vT       = (u16*)wsAlloc((size_t)BB * HH * HD * SS * 2);
    u16* o_bf     = (u16*)wsAlloc((size_t)BB * SS * EE * 2);
    int* covered  = (int*)wsAlloc((size_t)BB * SS * 4);

    const int M = BB * SS;   // 8192

    // f32 -> bf16 conversions
    f32_to_bf16_kernel<<<(BB * SS * EE + 255) / 256, 256, 0, stream>>>(x, x_bf, BB * SS * EE);
    f32_to_bf16_kernel<<<(E3 * EE + 255) / 256, 256, 0, stream>>>(w_in, w_in_bf, E3 * EE);
    f32_to_bf16_kernel<<<(EE * EE + 255) / 256, 256, 0, stream>>>(w_out, w_out_bf, EE * EE);

    // qkv = x @ in_proj_w^T + b  -> bf16 [B*S, 3E]
    {
        dim3 grid(M / 128, E3 / 128);
        wmma_gemm_bf16_kernel<<<grid, 256, 0, stream>>>(x_bf, w_in_bf, b_in, qkv_bf,
                                                        M, E3, EE, 1);
    }

    // v -> vT [B,H,HD,S]
    transpose_v_kernel<<<(BB * HH * HD * SS + 255) / 256, 256, 0, stream>>>(qkv_bf, vT);

    // fused attention (writes attn_weights directly, no atomics to global)
    attention_kernel<<<BB * (SS / 16), 256, 0, stream>>>(qkv_bf, vT, attn_weights, o_bf);

    // attn_output = o @ out_proj_w^T + b  -> f32
    {
        dim3 grid(M / 128, EE / 128);
        wmma_gemm_bf16_kernel<<<grid, 256, 0, stream>>>(o_bf, w_out_bf, b_out, attn_output,
                                                        M, EE, EE, 0);
    }

    // sequential greedy clustering + outputs
    cluster_kernel<<<BB, SS, 0, stream>>>(attn_weights, cids, covered);
    clustered_out_kernel<<<(BB * SS * EE + 255) / 256, 256, 0, stream>>>(x, covered, clustered);
}
